// SHATTN_3994319585511
// MI455X (gfx1250) — matrix operand order repaired and checked by reading, not compiled, not run on hardware
//
#include <hip/hip_runtime.h>
#include <hip/hip_bf16.h>

typedef __attribute__((ext_vector_type(16))) _Float16 v16h;
typedef __attribute__((ext_vector_type(8)))  _Float16 v8h;
typedef __attribute__((ext_vector_type(8)))  float    v8f;

#define BATCH 4
#define SEQ   4096
#define DMODEL 1024
#define HEADD  128
#define ROWS   (BATCH * SEQ)          // 16384
#define NSPLIT 4
#define QSCALE 0.08838834764831845f   // 1/sqrt(128)

// ---------------- workspace layout (bytes) ----------------
// xh : f16 x                [ROWS][DMODEL]            33,554,432
// WT : f16 W^T (q,k,v)      [3][DMODEL][HEADD]           786,432
// qf : f16 scaled q + bias  [ROWS][HEADD]              4,194,304
// kT : f16 k transposed     [B][HEADD][SEQ]            4,194,304
// vf : f16 v + bias         [ROWS][HEADD]              4,194,304
// Op : fp32 partial O       [NSPLIT][ROWS][HEADD]     33,554,432
// Mp : fp32 partial m       [NSPLIT][ROWS]               262,144
// Lp : fp32 partial l       [NSPLIT][ROWS]               262,144
#define OFF_XH 0u
#define OFF_WT (33554432u)
#define OFF_QF (OFF_WT + 786432u)
#define OFF_KT (OFF_QF + 4194304u)
#define OFF_VF (OFF_KT + 4194304u)
#define OFF_OP (OFF_VF + 4194304u)
#define OFF_MP (OFF_OP + 33554432u)
#define OFF_LP (OFF_MP + 262144u)

// ---------------------------------------------------------------------------
// Kernel 1: x (fp32) -> f16, vectorized 4 elements/thread
// ---------------------------------------------------------------------------
__global__ void shattn_cvt_x(const float* __restrict__ x, _Float16* __restrict__ xh) {
    int idx = (blockIdx.x * blockDim.x + threadIdx.x) * 4;
    float4 f = *(const float4*)(x + idx);
    xh[idx + 0] = (_Float16)f.x; xh[idx + 1] = (_Float16)f.y;
    xh[idx + 2] = (_Float16)f.z; xh[idx + 3] = (_Float16)f.w;
}

// ---------------------------------------------------------------------------
// Kernel 2: Wq/Wk/Wv [H][D] fp32 -> f16 transposed [D][H]
// ---------------------------------------------------------------------------
__global__ void shattn_cvt_w(const float* __restrict__ Wq, const float* __restrict__ Wk,
                             const float* __restrict__ Wv, _Float16* __restrict__ WT) {
    int idx = blockIdx.x * blockDim.x + threadIdx.x;       // 0 .. 3*131072-1
    int m   = idx >> 17;
    int rem = idx & 131071;
    int h   = rem >> 10;                                   // 0..127
    int d   = rem & 1023;                                  // 0..1023
    const float* W = (m == 0) ? Wq : (m == 1) ? Wk : Wv;
    WT[m * (DMODEL * HEADD) + d * HEADD + h] = (_Float16)W[h * DMODEL + d];
}

// ---------------------------------------------------------------------------
// Kernel 3: QKV projection GEMM.  One wave = one 16x128 output strip
// (all 8 N-tiles of one matrix) -> A operand reused 8x per K-chunk.
// waves: mat(3) x rowtile(1024) = 3072, 8 waves/block -> 384 blocks.
// ---------------------------------------------------------------------------
__global__ void shattn_qkv(const _Float16* __restrict__ xh, const _Float16* __restrict__ WT,
                           const float* __restrict__ bq, const float* __restrict__ bk,
                           const float* __restrict__ bv, _Float16* __restrict__ qf,
                           _Float16* __restrict__ kT, _Float16* __restrict__ vf) {
    int wave = blockIdx.x * (blockDim.x >> 5) + (threadIdx.x >> 5);
    int lane = threadIdx.x & 31;
    int mat     = wave >> 10;                 // 0=q,1=k,2=v
    int rowtile = wave & 1023;
    int rowbase = rowtile * 16;

    int n  = lane & 15;
    int hi = lane >> 4;

    v8f acc[8];
#pragma unroll
    for (int ct = 0; ct < 8; ++ct) acc[ct] = (v8f){};

    const _Float16* wmat = WT + mat * (DMODEL * HEADD);
#pragma unroll 2
    for (int kk = 0; kk < DMODEL; kk += 32) {
        // A operand: x rows (m = lane&15), two contiguous 8-half K segments
        const _Float16* ap = xh + (size_t)(rowbase + n) * DMODEL + kk + hi * 8;
        v8h alo = *(const v8h*)ap;
        v8h ahi = *(const v8h*)(ap + 16);
        v16h a = __builtin_shufflevector(alo, ahi, 0,1,2,3,4,5,6,7,8,9,10,11,12,13,14,15);
        // B operands: lane&15 = head col, elements = k (d) in table order
        const _Float16* wrow = wmat + (size_t)(kk + hi * 8) * HEADD + n;
#pragma unroll
        for (int ct = 0; ct < 8; ++ct) {
            v16h bm;
            for (int s = 0; s < 16; ++s) bm[s] = wrow[(size_t)(s + (s >> 3) * 8) * HEADD + ct * 16];
            acc[ct] = __builtin_amdgcn_wmma_f32_16x16x32_f16(false, a, false, bm,
                                                             (short)0, acc[ct], false, false);
        }
    }

    const float* bias = (mat == 0) ? bq : (mat == 1) ? bk : bv;
#pragma unroll
    for (int ct = 0; ct < 8; ++ct) {
        int h = ct * 16 + n;
        float bv_ = bias[h];
#pragma unroll
        for (int r = 0; r < 8; ++r) {
            int row = rowbase + r + 8 * hi;
            float val = acc[ct][r] + bv_;
            if (mat == 0) {
                qf[(size_t)row * HEADD + h] = (_Float16)(val * QSCALE);
            } else if (mat == 1) {
                int b = row >> 12, t = row & (SEQ - 1);
                kT[(size_t)b * (HEADD * SEQ) + (size_t)h * SEQ + t] = (_Float16)val;
            } else {
                vf[(size_t)row * HEADD + h] = (_Float16)val;
            }
        }
    }
}

// ---------------------------------------------------------------------------
// Kernel 4: split-K causal flash attention.
// One wave = one (16-query tile, split). Split s takes key-blocks
// kbase = s*32, s*32+128, ...  (perfect load balance across the triangle).
// Emits unnormalized partial O + per-row (m, l).
// 4096 waves, 4 waves/block -> 1024 blocks.
// ---------------------------------------------------------------------------
__global__ void shattn_flash(const _Float16* __restrict__ qf, const _Float16* __restrict__ kT,
                             const _Float16* __restrict__ vf, float* __restrict__ Opart,
                             float* __restrict__ Mpart, float* __restrict__ Lpart) {
    __shared__ __align__(16) _Float16 lds[4][16 * 32];   // per-wave P staging

    int wid  = threadIdx.x >> 5;
    int wave = blockIdx.x * 4 + wid;             // 0..4095
    int lane = threadIdx.x & 31;
    int split = wave & (NSPLIT - 1);
    int qw    = wave >> 2;                       // 0..1023
    int b     = qw >> 8;
    int qtile = qw & 255;
    int qbase = qtile * 16;

    int n  = lane & 15;
    int hi = lane >> 4;

    // Q tile (pre-scaled): A-operand layout for the 4 K-chunks of 32
    v16h qa[4];
#pragma unroll
    for (int c = 0; c < 4; ++c) {
        const _Float16* qp = qf + (size_t)(b * SEQ + qbase + n) * HEADD + c * 32 + hi * 8;
        v8h lo = *(const v8h*)qp;
        v8h hv = *(const v8h*)(qp + 16);
        qa[c] = __builtin_shufflevector(lo, hv, 0,1,2,3,4,5,6,7,8,9,10,11,12,13,14,15);
    }

    v8f o[8];
#pragma unroll
    for (int t = 0; t < 8; ++t) o[t] = (v8f){};
    float mrow[8], lrow[8];
#pragma unroll
    for (int r = 0; r < 8; ++r) { mrow[r] = -3.0e38f; lrow[r] = 0.0f; }

    const _Float16* kbt = kT + (size_t)b * (HEADD * SEQ);
    const _Float16* vbt = vf + (size_t)b * SEQ * HEADD;
    _Float16* myLds = &lds[wid][0];
    int kend = qbase + 16;

    for (int kbase = split * 32; kbase < kend; kbase += 32 * NSPLIT) {
        // ---- S = Q K^T (two 16-key tiles) ----
        v8f sa = {}, sb = {};
#pragma unroll
        for (int c = 0; c < 4; ++c) {
            const _Float16* kp = kbt + (size_t)(c * 32 + hi * 8) * SEQ + kbase + n;
            v16h k0, k1;                          // lane&15 = key (kbase+n / +16), elements = d
            for (int s = 0; s < 16; ++s) {
                k0[s] = kp[(size_t)(s + (s >> 3) * 8) * SEQ];
                k1[s] = kp[(size_t)(s + (s >> 3) * 8) * SEQ + 16];
            }
            sa = __builtin_amdgcn_wmma_f32_16x16x32_f16(false, qa[c], false, k0,
                                                        (short)0, sa, false, false);
            sb = __builtin_amdgcn_wmma_f32_16x16x32_f16(false, qa[c], false, k1,
                                                        (short)0, sb, false, false);
        }

        // ---- causal mask + online softmax ----
        float pa[8], pb[8], alpha[8];
#pragma unroll
        for (int r = 0; r < 8; ++r) {
            int row = qbase + r + 8 * hi;
            pa[r] = (kbase + n)      <= row ? sa[r] : -3.0e38f;
            pb[r] = (kbase + 16 + n) <= row ? sb[r] : -3.0e38f;
        }
#pragma unroll
        for (int r = 0; r < 8; ++r) {
            float t = fmaxf(pa[r], pb[r]);
#pragma unroll
            for (int mset = 1; mset < 16; mset <<= 1)
                t = fmaxf(t, __shfl_xor(t, mset, 32));
            float mnew = fmaxf(mrow[r], t);
            alpha[r] = __expf(mrow[r] - mnew);
            pa[r] = __expf(pa[r] - mnew);
            pb[r] = __expf(pb[r] - mnew);
            float s = pa[r] + pb[r];
#pragma unroll
            for (int mset = 1; mset < 16; mset <<= 1)
                s += __shfl_xor(s, mset, 32);
            lrow[r] = lrow[r] * alpha[r] + s;
            mrow[r] = mnew;
        }
#pragma unroll
        for (int t = 0; t < 8; ++t)
#pragma unroll
            for (int r = 0; r < 8; ++r) o[t][r] *= alpha[r];

        // ---- reshape P (C-layout) -> A-layout via per-wave LDS ----
#pragma unroll
        for (int r = 0; r < 8; ++r) {
            int row = r + 8 * hi;
            myLds[row * 32 + n]      = (_Float16)pa[r];
            myLds[row * 32 + 16 + n] = (_Float16)pb[r];
        }
        asm volatile("s_wait_dscnt 0" ::: "memory");
        const _Float16* lp = myLds + n * 32 + hi * 8;
        v8h plo = *(const v8h*)lp;
        v8h phi = *(const v8h*)(lp + 16);
        v16h pA = __builtin_shufflevector(plo, phi, 0,1,2,3,4,5,6,7,8,9,10,11,12,13,14,15);

        // ---- O += P V ----
        const _Float16* vp0 = vbt + (size_t)(kbase + hi * 8) * HEADD + n;   // lane&15 = head col, elements = key
#pragma unroll
        for (int t = 0; t < 8; ++t) {
            v16h vb16;
            for (int s = 0; s < 16; ++s) vb16[s] = vp0[(size_t)(s + (s >> 3) * 8) * HEADD + t * 16];
            o[t] = __builtin_amdgcn_wmma_f32_16x16x32_f16(false, pA, false, vb16,
                                                          (short)0, o[t], false, false);
        }
    }

    // ---- store unnormalized partials ----
    size_t sbase = (size_t)split * ROWS;
#pragma unroll
    for (int t = 0; t < 8; ++t)
#pragma unroll
        for (int r = 0; r < 8; ++r) {
            size_t row = (size_t)(b * SEQ + qbase + r + 8 * hi);
            Opart[(sbase + row) * HEADD + t * 16 + n] = o[t][r];
        }
    if (n == 0) {
#pragma unroll
        for (int r = 0; r < 8; ++r) {
            size_t row = (size_t)(b * SEQ + qbase + r + 8 * hi);
            Mpart[sbase + row] = mrow[r];
            Lpart[sbase + row] = lrow[r];
        }
    }
}

// ---------------------------------------------------------------------------
// Kernel 5: combine split-K partials (log-sum-exp merge), write fp32 out.
// ---------------------------------------------------------------------------
__global__ void shattn_combine(const float* __restrict__ Opart, const float* __restrict__ Mpart,
                               const float* __restrict__ Lpart, float* __restrict__ out) {
    int idx = blockIdx.x * blockDim.x + threadIdx.x;   // 0 .. ROWS*HEADD-1
    int row = idx >> 7;
    float m0 = Mpart[row], m1 = Mpart[ROWS + row];
    float m2 = Mpart[2 * ROWS + row], m3 = Mpart[3 * ROWS + row];
    float ms = fmaxf(fmaxf(m0, m1), fmaxf(m2, m3));
    float w0 = __expf(m0 - ms), w1 = __expf(m1 - ms);
    float w2 = __expf(m2 - ms), w3 = __expf(m3 - ms);
    float den = w0 * Lpart[row] + w1 * Lpart[ROWS + row] +
                w2 * Lpart[2 * ROWS + row] + w3 * Lpart[3 * ROWS + row];
    float num = w0 * Opart[idx] + w1 * Opart[ROWS * HEADD + idx] +
                w2 * Opart[2 * ROWS * HEADD + idx] + w3 * Opart[3 * ROWS * HEADD + idx];
    out[idx] = num / den;
}

// ---------------------------------------------------------------------------
extern "C" void kernel_launch(void* const* d_in, const int* in_sizes, int n_in,
                              void* d_out, int out_size, void* d_ws, size_t ws_size,
                              hipStream_t stream) {
    const float* x  = (const float*)d_in[0];
    const float* Wq = (const float*)d_in[1];
    const float* bq = (const float*)d_in[2];
    const float* Wk = (const float*)d_in[3];
    const float* bk = (const float*)d_in[4];
    const float* Wv = (const float*)d_in[5];
    const float* bv = (const float*)d_in[6];
    float* out = (float*)d_out;

    char* ws = (char*)d_ws;
    _Float16* xh = (_Float16*)(ws + OFF_XH);
    _Float16* WT = (_Float16*)(ws + OFF_WT);
    _Float16* qf = (_Float16*)(ws + OFF_QF);
    _Float16* kT = (_Float16*)(ws + OFF_KT);
    _Float16* vf = (_Float16*)(ws + OFF_VF);
    float*    Op = (float*)(ws + OFF_OP);
    float*    Mp = (float*)(ws + OFF_MP);
    float*    Lp = (float*)(ws + OFF_LP);

    // 1) x -> f16 : 16,777,216 elems, 4/thread
    shattn_cvt_x<<<16384, 256, 0, stream>>>(x, xh);
    // 2) weights -> f16 transposed : 393,216 elems
    shattn_cvt_w<<<1536, 256, 0, stream>>>(Wq, Wk, Wv, WT);
    // 3) QKV GEMM : 3072 wave-strips (16x128 each), 8 waves per block
    shattn_qkv<<<384, 256, 0, stream>>>(xh, WT, bq, bk, bv, qf, kT, vf);
    // 4) split-K flash attention : 4096 waves, 4 per block
    shattn_flash<<<1024, 128, 0, stream>>>(qf, kT, vf, Op, Mp, Lp);
    // 5) combine partials : 2,097,152 elems
    shattn_combine<<<8192, 256, 0, stream>>>(Op, Mp, Lp, out);
}
